// WorldModel_43705587204219
// MI455X (gfx1250) — matrix operand order repaired and checked by reading, not compiled, not run on hardware
//
#include <hip/hip_runtime.h>
#include <hip/hip_bf16.h>
#include <math.h>

// ---------------------------------------------------------------------------
// WorldModel forward loss on gfx1250 (MI455X).  All dense math (convs via
// implicit GEMM, conv-transpose via fractional-stride gather, linear layers)
// goes through v_wmma_f32_16x16x32_f16.  Each wave computes a 32x32 C macro-
// tile: 4 accumulators, 4 WMMAs per 32-wide K chunk.  Weight tiles (f16
// [N][K]) are streamed global->LDS by the Tensor Data Mover (TDM) with D#
// padding producing pitch-40 rows; the wave's vector pipe concurrently does
// the im2col gather + f32->f16 convert for the A tile, then joins with
// s_wait_tensorcnt.  Fragment loads are 2x ds_load_b128 each.
// ---------------------------------------------------------------------------

typedef __attribute__((ext_vector_type(16))) _Float16 v16h;
typedef __attribute__((ext_vector_type(8)))  float    v8f;
typedef __attribute__((ext_vector_type(4)))  unsigned v4u;
typedef __attribute__((ext_vector_type(4)))  int      v4i;
typedef __attribute__((ext_vector_type(8)))  int      v8i;

#define BB 16
#define TT 64
#define SD 256
#define LD 256
#define CC 32
#define KK_CAT 32
#define AA 6
#define FS 1280          // SD + C*K
#define NIMG_ENC 1040    // B*(T+1)
#define NIMG_DEC 1024    // B*T
#define PITCH 40         // LDS row pitch in halves (80B = 64B tile row + 16B pad)

#if __has_builtin(__builtin_amdgcn_tensor_load_to_lds)
#define HAS_TDM 1
#else
#define HAS_TDM 0
#endif

#if HAS_TDM
// ---------------------------------------------------------------------------
// TDM: DMA a 32x32-element f16 tile from a row-major [rows][ldK] f16 matrix
// into LDS at ldsOff, padding each 64B row with 16B -> 80B pitch (PITCH=40).
// remRows/remK clamp via D# tensor dims (OOB reads return zero).
// D# layout per CDNA5 ISA.
// ---------------------------------------------------------------------------
__device__ __forceinline__ void tdm_load_tile(const _Float16* gptr, unsigned ldsOff,
                                              int remRows, int remK, int ldK) {
  unsigned long long ga = (unsigned long long)(size_t)gptr;
  v4u g0;
  g0[0] = 1u;                                                 // count=1, user D#
  g0[1] = ldsOff;                                             // lds_addr (bytes)
  g0[2] = (unsigned)(ga & 0xFFFFFFFFu);                       // global_addr[31:0]
  g0[3] = (unsigned)((ga >> 32) & 0x01FFFFFFu) | (1u << 31);  // ga[56:32] | type=2
  unsigned td0 = (unsigned)remK;     // tensor_dim0 (elements in a row)
  unsigned td1 = (unsigned)remRows;  // tensor_dim1 (rows)
  unsigned long long st0 = (unsigned long long)(unsigned)ldK; // dim0 stride (elems)
  v8i g1;
  g1[0] = (int)((1u << 16)      // data_size = 2 bytes
              | (1u << 20)      // pad_enable
              | (3u << 22)      // pad_interval: 16 DWORDs (64B) before pad
              | (3u << 25));    // pad_amount: 4 DWORDs (16B)
  g1[1] = (int)((td0 & 0xFFFFu) << 16);                             // dim0[15:0]
  g1[2] = (int)(((td0 >> 16) & 0xFFFFu) | ((td1 & 0xFFFFu) << 16)); // dim0[31:16], dim1[15:0]
  g1[3] = (int)(((td1 >> 16) & 0xFFFFu) | (32u << 16));             // dim1[31:16], tile_dim0=32
  g1[4] = (int)32u;                                                 // tile_dim1=32
  g1[5] = (int)(unsigned)(st0 & 0xFFFFFFFFull);                     // stride0[31:0]
  g1[6] = (int)(unsigned)((st0 >> 32) & 0xFFFFull);                 // stride0[47:32]
  g1[7] = 0;
  v4i z4 = {0, 0, 0, 0};
#if defined(__clang_major__) && __clang_major__ >= 23
  v8i z8 = {0, 0, 0, 0, 0, 0, 0, 0};
  __builtin_amdgcn_tensor_load_to_lds(g0, g1, z4, z4, z8, 0);
#else
  __builtin_amdgcn_tensor_load_to_lds(g0, g1, z4, z4, 0);
#endif
}
#endif  // HAS_TDM

// ---------------------------------------------------------------------------
// Fragment loads per CDNA5 ISA 7.12.2 (16-bit A 16x32 / B 32x16, wave32).
// As is [32 rows][K=32] halves (pitch 40); Bs is [32 cols][K=32] (pitch 40).
// Both are K-contiguous per lane -> 2x ds_load_b128 each.
// ---------------------------------------------------------------------------
__device__ __forceinline__ v16h load_a(const _Float16* As, int mi) {
  const int lane = threadIdx.x & 31;
  const int hf = lane >> 4, r = lane & 15;
  v16h a;
#pragma unroll
  for (int j = 0; j < 8; ++j) {
    a[j]     = As[(mi * 16 + r) * PITCH + hf * 8 + j];        // K 0..15 region
    a[j + 8] = As[(mi * 16 + r) * PITCH + 16 + hf * 8 + j];   // K 16..31 region
  }
  return a;
}
__device__ __forceinline__ v16h load_b(const _Float16* Bs, int ni) {
  const int lane = threadIdx.x & 31;
  const int hf = lane >> 4, r = lane & 15;
  v16h b;
#pragma unroll
  for (int j = 0; j < 8; ++j) {
    b[j]     = Bs[(ni * 16 + r) * PITCH + hf * 8 + j];
    b[j + 8] = Bs[(ni * 16 + r) * PITCH + 16 + hf * 8 + j];
  }
  return b;
}
#define WMMA_F16(a, b, c) \
  __builtin_amdgcn_wmma_f32_16x16x32_f16(false, (a), false, (b), (short)0, (c), false, false)

// Stage a 32x32 weight tile from f16 [N][K] into Bs (TDM if available).
__device__ __forceinline__ void stage_weights(_Float16* Bs, const _Float16* Wt,
                                              int n0, int k0, int N, int K) {
#if HAS_TDM
  tdm_load_tile(Wt + (size_t)n0 * K + k0, (unsigned)(size_t)Bs, N - n0, K - k0, K);
#else
  const int lane = threadIdx.x & 31;
  if (k0 + 32 < K && n0 + lane < N)
    __builtin_prefetch(Wt + (size_t)(n0 + lane) * K + k0 + 32, 0, 1);
  for (int i = lane; i < 1024; i += 32) {
    int c = n0 + (i >> 5), kk = k0 + (i & 31);
    Bs[(i >> 5) * PITCH + (i & 31)] =
        (c < N && kk < K) ? Wt[(size_t)c * K + kk] : (_Float16)0.f;
  }
#endif
}
__device__ __forceinline__ void join_weights() {
#if HAS_TDM
  __builtin_amdgcn_s_wait_tensorcnt((short)0);
#endif
}

// ---------------------------------------------------------------------------
// Generic GEMM: C[M,N] = act(A_f32[M,K](lda) * W_f16[N][K] + bias)
// One wave per 32x32 macro-tile.
// ---------------------------------------------------------------------------
__global__ __launch_bounds__(32) void k_gemm(const float* __restrict__ A, int lda,
                                             const _Float16* __restrict__ Bw,
                                             const float* __restrict__ bias,
                                             float* __restrict__ C, int ldc,
                                             int M, int N, int K, int relu) {
  __shared__ _Float16 As[32 * PITCH];
  __shared__ _Float16 Bs[32 * PITCH];
  const int lane = threadIdx.x & 31;
  const int m0 = blockIdx.x * 32;
  const int n0 = blockIdx.y * 32;
  v8f acc[2][2] = {};
  for (int k0 = 0; k0 < K; k0 += 32) {
    stage_weights(Bs, Bw, n0, k0, N, K);         // TDM streams weights to LDS
    if (k0 + 32 < K && m0 + lane < M)
      __builtin_prefetch(A + (size_t)(m0 + lane) * lda + k0 + 32, 0, 1);
    for (int i = lane; i < 1024; i += 32) {      // overlap: gather+convert A
      int m = m0 + (i >> 5), kk = k0 + (i & 31);
      float v = (m < M && kk < K) ? A[(size_t)m * lda + kk] : 0.f;
      As[(i >> 5) * PITCH + (i & 31)] = (_Float16)v;
    }
    join_weights();                              // s_wait_tensorcnt 0
    __syncthreads();
    v16h a0 = load_a(As, 0), a1 = load_a(As, 1);
    v16h b0 = load_b(Bs, 0), b1 = load_b(Bs, 1);
    acc[0][0] = WMMA_F16(a0, b0, acc[0][0]);
    acc[0][1] = WMMA_F16(a0, b1, acc[0][1]);
    acc[1][0] = WMMA_F16(a1, b0, acc[1][0]);
    acc[1][1] = WMMA_F16(a1, b1, acc[1][1]);
    __syncthreads();
  }
  const int hf = lane >> 4, r = lane & 15;
#pragma unroll
  for (int ni = 0; ni < 2; ++ni) {
    int col = n0 + ni * 16 + r;
    if (col >= N) continue;
    float bb = bias ? bias[col] : 0.f;
#pragma unroll
    for (int mi = 0; mi < 2; ++mi) {
#pragma unroll
      for (int v = 0; v < 8; ++v) {
        int row = m0 + mi * 16 + v + 8 * hf;
        if (row < M) {
          float x = acc[mi][ni][v] + bb;
          if (relu) x = fmaxf(x, 0.f);
          C[(size_t)row * ldc + col] = x;
        }
      }
    }
  }
}

// ---------------------------------------------------------------------------
// Implicit-GEMM stride-2 k4 pad1 conv: M = Nimg*OH*OW rows, N = Cout.
// Wt is f16 [Cout][K=Cin*16] (natural torch layout).  Output NCHW f32 + ReLU.
// ---------------------------------------------------------------------------
__global__ __launch_bounds__(32) void k_conv(const float* __restrict__ in,
                                             const _Float16* __restrict__ Wt,
                                             const float* __restrict__ bias,
                                             float* __restrict__ out,
                                             int Nimg, int Cin, int IH, int IW,
                                             int Cout, int OH, int OW) {
  __shared__ _Float16 As[32 * PITCH];
  __shared__ _Float16 Bs[32 * PITCH];
  const int lane = threadIdx.x & 31;
  const int m0 = blockIdx.x * 32;
  const int n0 = blockIdx.y * 32;
  const int M = Nimg * OH * OW;
  const int K = Cin * 16;
  v8f acc[2][2] = {};
  for (int k0 = 0; k0 < K; k0 += 32) {
    stage_weights(Bs, Wt, n0, k0, Cout, K);      // TDM streams weights to LDS
    for (int i = lane; i < 1024; i += 32) {      // overlap: im2col gather
      int m = m0 + (i >> 5), kk = k0 + (i & 31);
      float v = 0.f;
      if (m < M && kk < K) {
        int n = m / (OH * OW), p = m - n * OH * OW;
        int oy = p / OW, ox = p - oy * OW;
        int ci = kk >> 4, ky = (kk >> 2) & 3, kx = kk & 3;
        int iy = oy * 2 - 1 + ky, ix = ox * 2 - 1 + kx;
        if (iy >= 0 && iy < IH && ix >= 0 && ix < IW)
          v = in[(((size_t)n * Cin + ci) * IH + iy) * IW + ix];
      }
      As[(i >> 5) * PITCH + (i & 31)] = (_Float16)v;
    }
    join_weights();
    __syncthreads();
    v16h a0 = load_a(As, 0), a1 = load_a(As, 1);
    v16h b0 = load_b(Bs, 0), b1 = load_b(Bs, 1);
    acc[0][0] = WMMA_F16(a0, b0, acc[0][0]);
    acc[0][1] = WMMA_F16(a0, b1, acc[0][1]);
    acc[1][0] = WMMA_F16(a1, b0, acc[1][0]);
    acc[1][1] = WMMA_F16(a1, b1, acc[1][1]);
    __syncthreads();
  }
  const int hf = lane >> 4, r = lane & 15;
#pragma unroll
  for (int ni = 0; ni < 2; ++ni) {
    int col = n0 + ni * 16 + r;
    if (col >= Cout) continue;
    float bb = bias[col];
#pragma unroll
    for (int mi = 0; mi < 2; ++mi) {
#pragma unroll
      for (int v = 0; v < 8; ++v) {
        int row = m0 + mi * 16 + v + 8 * hf;
        if (row < M) {
          int n = row / (OH * OW), p = row - n * OH * OW;
          int oy = p / OW, ox = p - oy * OW;
          float x = fmaxf(acc[mi][ni][v] + bb, 0.f);
          out[(((size_t)n * Cout + col) * OH + oy) * OW + ox] = x;
        }
      }
    }
  }
}

// ---------------------------------------------------------------------------
// Implicit-GEMM stride-2 VALID conv-transpose (kernel KK).  OH=(IH-1)*2+KK.
// Wt is f16 [Cout][K=Cin*KK*KK].
// ---------------------------------------------------------------------------
__global__ __launch_bounds__(32) void k_convT(const float* __restrict__ in,
                                              const _Float16* __restrict__ Wt,
                                              const float* __restrict__ bias,
                                              float* __restrict__ out,
                                              int Nimg, int Cin, int IH, int IW,
                                              int Cout, int KKk, int OH, int OW,
                                              int relu) {
  __shared__ _Float16 As[32 * PITCH];
  __shared__ _Float16 Bs[32 * PITCH];
  const int lane = threadIdx.x & 31;
  const int m0 = blockIdx.x * 32;
  const int n0 = blockIdx.y * 32;
  const int M = Nimg * OH * OW;
  const int K = Cin * KKk * KKk;
  v8f acc[2][2] = {};
  for (int k0 = 0; k0 < K; k0 += 32) {
    stage_weights(Bs, Wt, n0, k0, Cout, K);      // TDM streams weights to LDS
    for (int i = lane; i < 1024; i += 32) {      // overlap: frac-stride gather
      int m = m0 + (i >> 5), kk = k0 + (i & 31);
      float v = 0.f;
      if (m < M && kk < K) {
        int n = m / (OH * OW), p = m - n * OH * OW;
        int oy = p / OW, ox = p - oy * OW;
        int ci = kk / (KKk * KKk), rr = kk - ci * KKk * KKk;
        int ky = rr / KKk, kx = rr - ky * KKk;
        int ty = oy + ky - (KKk - 1), tx = ox + kx - (KKk - 1);
        if (ty >= 0 && tx >= 0 && !(ty & 1) && !(tx & 1)) {
          int iy = ty >> 1, ix = tx >> 1;
          if (iy < IH && ix < IW)
            v = in[(((size_t)n * Cin + ci) * IH + iy) * IW + ix];
        }
      }
      As[(i >> 5) * PITCH + (i & 31)] = (_Float16)v;
    }
    join_weights();
    __syncthreads();
    v16h a0 = load_a(As, 0), a1 = load_a(As, 1);
    v16h b0 = load_b(Bs, 0), b1 = load_b(Bs, 1);
    acc[0][0] = WMMA_F16(a0, b0, acc[0][0]);
    acc[0][1] = WMMA_F16(a0, b1, acc[0][1]);
    acc[1][0] = WMMA_F16(a1, b0, acc[1][0]);
    acc[1][1] = WMMA_F16(a1, b1, acc[1][1]);
    __syncthreads();
  }
  const int hf = lane >> 4, r = lane & 15;
#pragma unroll
  for (int ni = 0; ni < 2; ++ni) {
    int col = n0 + ni * 16 + r;
    if (col >= Cout) continue;
    float bb = bias[col];
#pragma unroll
    for (int mi = 0; mi < 2; ++mi) {
#pragma unroll
      for (int v = 0; v < 8; ++v) {
        int row = m0 + mi * 16 + v + 8 * hf;
        if (row < M) {
          int n = row / (OH * OW), p = row - n * OH * OW;
          int oy = p / OW, ox = p - oy * OW;
          float x = acc[mi][ni][v] + bb;
          if (relu) x = fmaxf(x, 0.f);
          out[(((size_t)n * Cout + col) * OH + oy) * OW + ox] = x;
        }
      }
    }
  }
}

// ------------------------- small utility kernels ---------------------------
__global__ void k_cvt_plain(const float* __restrict__ s, _Float16* __restrict__ d, int n) {
  int i = blockIdx.x * blockDim.x + threadIdx.x;
  if (i < n) d[i] = (_Float16)s[i];
}
// src [R, C] row-major -> dst [C, R] f16 (linear weights [K,N] -> [N,K])
__global__ void k_cvt_t(const float* __restrict__ s, _Float16* __restrict__ d, int R, int C) {
  int i = blockIdx.x * blockDim.x + threadIdx.x;
  if (i < R * C) { int r = i / C, c = i - r * C; d[(size_t)c * R + r] = (_Float16)s[i]; }
}
__global__ void k_zero(float* p, int n) {
  for (int i = blockIdx.x * blockDim.x + threadIdx.x; i < n; i += gridDim.x * blockDim.x)
    p[i] = 0.f;
}
__global__ void k_concat_sa(const float* __restrict__ stoch, const int* __restrict__ act,
                            int t, float* __restrict__ xcat) {
  int i = blockIdx.x * blockDim.x + threadIdx.x;
  if (i >= BB * 1030) return;
  int b = i / 1030, c = i - b * 1030;
  float v;
  if (c < 1024) v = stoch[b * 1024 + c];
  else { int a = act[b * TT + t]; v = (c - 1024 == a) ? 1.f : 0.f; }
  xcat[i] = v;
}
__global__ void k_concat_he(const float* __restrict__ h, const float* __restrict__ emb,
                            int t, float* __restrict__ he) {
  int i = blockIdx.x * blockDim.x + threadIdx.x;
  if (i >= BB * 512) return;
  int b = i >> 9, c = i & 511;
  he[i] = (c < 256) ? h[b * 256 + c]
                    : emb[((size_t)b * (TT + 1) + t + 1) * LD + (c - 256)];
}
__global__ void k_gru(const float* __restrict__ gi, const float* __restrict__ gh,
                      float* __restrict__ h) {
  int i = blockIdx.x * blockDim.x + threadIdx.x;
  if (i >= BB * SD) return;
  int b = i >> 8, j = i & 255;
  float ir = gi[b * 768 + j],       hr = gh[b * 768 + j];
  float iz = gi[b * 768 + 256 + j], hz = gh[b * 768 + 256 + j];
  float in = gi[b * 768 + 512 + j], hn = gh[b * 768 + 512 + j];
  float r = 1.f / (1.f + expf(-(ir + hr)));
  float z = 1.f / (1.f + expf(-(iz + hz)));
  float n = tanhf(in + r * hn);
  h[i] = (1.f - z) * n + z * h[i];
}
// straight-through categorical sample: forward value is one-hot(argmax(post+g))
__global__ void k_sample(const float* __restrict__ post_t, const float* __restrict__ gum,
                         int t, float* __restrict__ stoch) {
  int i = blockIdx.x * blockDim.x + threadIdx.x;
  if (i >= BB * CC) return;
  int b = i >> 5, c = i & 31;
  const float* lg = post_t + b * 1024 + c * 32;
  const float* g  = gum + (((size_t)t * BB + b) * CC + c) * KK_CAT;
  int best = 0; float bv = -1e30f;
  for (int k = 0; k < 32; ++k) { float v = lg[k] + g[k]; if (v > bv) { bv = v; best = k; } }
  for (int k = 0; k < 32; ++k) stoch[b * 1024 + c * 32 + k] = (k == best) ? 1.f : 0.f;
}
__global__ void k_write_full(const float* __restrict__ h, const float* __restrict__ stoch,
                             int t, float* __restrict__ fulls) {
  int i = blockIdx.x * blockDim.x + threadIdx.x;
  if (i >= BB * FS) return;
  int b = i / FS, c = i - b * FS;
  fulls[((size_t)b * TT + t) * FS + c] = (c < 256) ? h[b * 256 + c]
                                                   : stoch[b * 1024 + (c - 256)];
}
// ------------------------------ losses -------------------------------------
__global__ void k_mse_recon(const float* __restrict__ recon, const float* __restrict__ obs,
                            float* __restrict__ accum) {
  __shared__ float red[256];
  const long total = (long)NIMG_DEC * 3 * 4096;
  float s = 0.f;
  for (long i = (long)blockIdx.x * blockDim.x + threadIdx.x; i < total;
       i += (long)gridDim.x * blockDim.x) {
    long n = i / 12288, rem = i - n * 12288;
    long b = n >> 6, t = n & 63;
    float d = recon[i] - obs[(b * (TT + 1) + t + 1) * 12288 + rem];
    s += d * d;
  }
  red[threadIdx.x] = s; __syncthreads();
  for (int o = 128; o > 0; o >>= 1) {
    if ((int)threadIdx.x < o) red[threadIdx.x] += red[threadIdx.x + o];
    __syncthreads();
  }
  if (threadIdx.x == 0) atomicAdd(accum, red[0] / (float)total);
}
__global__ void k_mse_reward(const float* __restrict__ rp, const float* __restrict__ rw,
                             float* __restrict__ accum) {
  __shared__ float red[256];
  float s = 0.f;
  int i = blockIdx.x * blockDim.x + threadIdx.x;
  if (i < NIMG_DEC) { float d = rp[i] - rw[i]; s = d * d; }  // row b*T+t matches rewards[b,t]
  red[threadIdx.x] = s; __syncthreads();
  for (int o = 128; o > 0; o >>= 1) {
    if ((int)threadIdx.x < o) red[threadIdx.x] += red[threadIdx.x + o];
    __syncthreads();
  }
  if (threadIdx.x == 0) atomicAdd(accum + 1, red[0] / (float)NIMG_DEC);
}
__global__ void k_kl(const float* __restrict__ priors, const float* __restrict__ posts,
                     float* __restrict__ accum) {
  __shared__ float red[256];
  const int NE = TT * BB * CC;  // 32768 rows of K=32 logits
  float s = 0.f;
  for (int e = blockIdx.x * blockDim.x + threadIdx.x; e < NE; e += gridDim.x * blockDim.x) {
    const float* p = posts + (size_t)e * 32;
    const float* q = priors + (size_t)e * 32;
    float pm = -1e30f, qm = -1e30f;
    for (int k = 0; k < 32; ++k) { pm = fmaxf(pm, p[k]); qm = fmaxf(qm, q[k]); }
    float ps = 0.f, qs = 0.f;
    for (int k = 0; k < 32; ++k) { ps += expf(p[k] - pm); qs += expf(q[k] - qm); }
    float v = 0.f;
    for (int k = 0; k < 32; ++k) {
      float pp = expf(p[k] - pm) / ps + 1e-8f;
      float qp = expf(q[k] - qm) / qs + 1e-8f;
      v += pp * (logf(pp) - logf(qp));  // KL_BAL*x + (1-KL_BAL)*x == x in fwd
    }
    s += v;
  }
  red[threadIdx.x] = s; __syncthreads();
  for (int o = 128; o > 0; o >>= 1) {
    if ((int)threadIdx.x < o) red[threadIdx.x] += red[threadIdx.x + o];
    __syncthreads();
  }
  if (threadIdx.x == 0) atomicAdd(accum + 2, red[0] / (float)NE);
}
__global__ void k_final(const float* __restrict__ accum, float* __restrict__ out) {
  if (threadIdx.x == 0 && blockIdx.x == 0) out[0] = accum[0] + accum[1] + accum[2];
}

// ---------------------------------------------------------------------------
extern "C" void kernel_launch(void* const* d_in, const int* in_sizes, int n_in,
                              void* d_out, int out_size, void* d_ws, size_t ws_size,
                              hipStream_t stream) {
  (void)in_sizes; (void)n_in; (void)out_size; (void)ws_size;
  const float* obs    = (const float*)d_in[0];
  const int*   act    = (const int*)d_in[1];
  const float* rew    = (const float*)d_in[2];
  const float* ew1 = (const float*)d_in[3],  *eb1 = (const float*)d_in[4];
  const float* ew2 = (const float*)d_in[5],  *eb2 = (const float*)d_in[6];
  const float* ew3 = (const float*)d_in[7],  *eb3 = (const float*)d_in[8];
  const float* ew4 = (const float*)d_in[9],  *eb4 = (const float*)d_in[10];
  const float* enc_lw = (const float*)d_in[11], *enc_lb = (const float*)d_in[12];
  const float* pre_w  = (const float*)d_in[13], *pre_b  = (const float*)d_in[14];
  const float* gru_wi = (const float*)d_in[15], *gru_wh = (const float*)d_in[16];
  const float* gru_bi = (const float*)d_in[17], *gru_bh = (const float*)d_in[18];
  const float* pri_w1 = (const float*)d_in[19], *pri_b1 = (const float*)d_in[20];
  const float* pri_w2 = (const float*)d_in[21], *pri_b2 = (const float*)d_in[22];
  const float* post_w1 = (const float*)d_in[23], *post_b1 = (const float*)d_in[24];
  const float* post_w2 = (const float*)d_in[25], *post_b2 = (const float*)d_in[26];
  const float* dec_lw = (const float*)d_in[27], *dec_lb = (const float*)d_in[28];
  const float* dw1 = (const float*)d_in[29], *db1 = (const float*)d_in[30];
  const float* dw2 = (const float*)d_in[31], *db2 = (const float*)d_in[32];
  const float* dw3 = (const float*)d_in[33], *db3 = (const float*)d_in[34];
  const float* dw4 = (const float*)d_in[35], *db4 = (const float*)d_in[36];
  const float* rw1 = (const float*)d_in[37], *rb1 = (const float*)d_in[38];
  const float* rw2 = (const float*)d_in[39], *rb2 = (const float*)d_in[40];
  const float* gum = (const float*)d_in[41];
  float* out = (float*)d_out;

  // ---- workspace bump allocator (fixed, deterministic layout) ----
  char* wsp = (char*)d_ws;
  auto alloc = [&](size_t bytes) -> char* {
    char* p = wsp; wsp += (bytes + 255) & ~(size_t)255; return p;
  };
  auto fh = [&](size_t elems) { return (_Float16*)alloc(elems * 2); };
  auto ff = [&](size_t elems) { return (float*)alloc(elems * 4); };

  // all weights stored f16 [N][K]
  _Float16* w_enclw = fh(4096 * 256);
  _Float16* w_pre   = fh(1030 * 256);
  _Float16* w_gwi   = fh(256 * 768);
  _Float16* w_gwh   = fh(256 * 768);
  _Float16* w_pri1  = fh(256 * 256);
  _Float16* w_pri2  = fh(256 * 1024);
  _Float16* w_post1 = fh(512 * 256);
  _Float16* w_post2 = fh(256 * 1024);
  _Float16* w_declw = fh(1280 * 1024);
  _Float16* w_rw1   = fh(1280 * 256);
  _Float16* w_rw2   = fh(256 * 1);
  _Float16* w_ew1   = fh(32 * 48);
  _Float16* w_ew2   = fh(64 * 512);
  _Float16* w_ew3   = fh(128 * 1024);
  _Float16* w_ew4   = fh(256 * 2048);
  _Float16* w_dw1   = fh((size_t)128 * 25600);
  _Float16* w_dw2   = fh(64 * 3200);
  _Float16* w_dw3   = fh(32 * 2304);
  _Float16* w_dw4   = fh(3 * 1152);

  float* bufA   = ff(34078720);  // act1 / act4 / d1 / d3
  float* bufB   = ff(17039360);  // act2 / d2 / recon
  float* bufC   = ff(8519680);   // act3 / d0
  float* embeds = ff((size_t)NIMG_ENC * LD);
  float* fulls  = ff((size_t)NIMG_DEC * FS);
  float* priors = ff((size_t)TT * BB * 1024);
  float* posts  = ff((size_t)TT * BB * 1024);
  float* h      = ff(BB * SD);
  float* stoch  = ff(BB * 1024);
  float* xcat   = ff(BB * 1030);
  float* xin    = ff(BB * SD);
  float* gi     = ff(BB * 768);
  float* gh     = ff(BB * 768);
  float* t1     = ff(BB * 256);
  float* he     = ff(BB * 512);
  float* t2     = ff(BB * 256);
  float* rtmp   = ff((size_t)NIMG_DEC * 256);
  float* rpred  = ff(NIMG_DEC);
  float* accum  = ff(4);

  auto cvt = [&](const float* s, _Float16* d, size_t n) {
    k_cvt_plain<<<(unsigned)((n + 255) / 256), 256, 0, stream>>>(s, d, (int)n);
  };
  auto cvtT = [&](const float* s, _Float16* d, int R, int C) {  // [R,C] -> [C,R]
    k_cvt_t<<<(R * C + 255) / 256, 256, 0, stream>>>(s, d, R, C);
  };
  auto gemm = [&](const float* A, int lda, const _Float16* Bw, const float* bias,
                  float* C, int ldc, int M, int N, int K, int relu) {
    dim3 g((M + 31) / 32, (N + 31) / 32);
    k_gemm<<<g, 32, 0, stream>>>(A, lda, Bw, bias, C, ldc, M, N, K, relu);
  };
  auto conv = [&](const float* in, const _Float16* Wt, const float* bias, float* o,
                  int Nimg, int Cin, int IH, int IW, int Cout, int OH, int OW) {
    dim3 g((Nimg * OH * OW + 31) / 32, (Cout + 31) / 32);
    k_conv<<<g, 32, 0, stream>>>(in, Wt, bias, o, Nimg, Cin, IH, IW, Cout, OH, OW);
  };
  auto convT = [&](const float* in, const _Float16* Wt, const float* bias, float* o,
                   int Nimg, int Cin, int IH, int IW, int Cout, int KKk, int OH, int OW,
                   int relu) {
    dim3 g((Nimg * OH * OW + 31) / 32, (Cout + 31) / 32);
    k_convT<<<g, 32, 0, stream>>>(in, Wt, bias, o, Nimg, Cin, IH, IW, Cout, KKk, OH, OW, relu);
  };

  // ---- weight conversion to f16 [N][K] ----
  // conv / convT weights are already [O][K] row-major -> plain convert
  cvt(ew1, w_ew1, 32 * 48);
  cvt(ew2, w_ew2, (size_t)64 * 512);
  cvt(ew3, w_ew3, (size_t)128 * 1024);
  cvt(ew4, w_ew4, (size_t)256 * 2048);
  cvt(dw1, w_dw1, (size_t)128 * 25600);
  cvt(dw2, w_dw2, (size_t)64 * 3200);
  cvt(dw3, w_dw3, (size_t)32 * 2304);
  cvt(dw4, w_dw4, (size_t)3 * 1152);
  // linear weights are [K,N] -> transpose to [N,K]
  cvtT(enc_lw, w_enclw, 4096, 256);
  cvtT(pre_w,  w_pre,   1030, 256);
  cvtT(gru_wi, w_gwi,   256, 768);
  cvtT(gru_wh, w_gwh,   256, 768);
  cvtT(pri_w1, w_pri1,  256, 256);
  cvtT(pri_w2, w_pri2,  256, 1024);
  cvtT(post_w1, w_post1, 512, 256);
  cvtT(post_w2, w_post2, 256, 1024);
  cvtT(dec_lw, w_declw, 1280, 1024);
  cvtT(rw1, w_rw1, 1280, 256);
  cvtT(rw2, w_rw2, 256, 1);

  // ---- state / accumulator init ----
  k_zero<<<16, 256, 0, stream>>>(h, BB * SD);
  k_zero<<<64, 256, 0, stream>>>(stoch, BB * 1024);
  k_zero<<<1, 256, 0, stream>>>(accum, 4);

  // ---- encoder: 4 implicit-GEMM convs + linear ----
  conv(obs,  w_ew1, eb1, bufA, NIMG_ENC,   3, 64, 64,  32, 32, 32);
  conv(bufA, w_ew2, eb2, bufB, NIMG_ENC,  32, 32, 32,  64, 16, 16);
  conv(bufB, w_ew3, eb3, bufC, NIMG_ENC,  64, 16, 16, 128,  8,  8);
  conv(bufC, w_ew4, eb4, bufA, NIMG_ENC, 128,  8,  8, 256,  4,  4);
  gemm(bufA, 4096, w_enclw, enc_lb, embeds, LD, NIMG_ENC, LD, 4096, 0);

  // ---- RSSM scan ----
  for (int t = 0; t < TT; ++t) {
    k_concat_sa<<<(BB * 1030 + 255) / 256, 256, 0, stream>>>(stoch, act, t, xcat);
    gemm(xcat, 1030, w_pre, pre_b, xin, 256, BB, 256, 1030, 1);
    gemm(xin, 256, w_gwi, gru_bi, gi, 768, BB, 768, 256, 0);
    gemm(h, 256, w_gwh, gru_bh, gh, 768, BB, 768, 256, 0);
    k_gru<<<(BB * SD + 255) / 256, 256, 0, stream>>>(gi, gh, h);
    gemm(h, 256, w_pri1, pri_b1, t1, 256, BB, 256, 256, 1);
    gemm(t1, 256, w_pri2, pri_b2, priors + (size_t)t * BB * 1024, 1024, BB, 1024, 256, 0);
    k_concat_he<<<(BB * 512 + 255) / 256, 256, 0, stream>>>(h, embeds, t, he);
    gemm(he, 512, w_post1, post_b1, t2, 256, BB, 256, 512, 1);
    gemm(t2, 256, w_post2, post_b2, posts + (size_t)t * BB * 1024, 1024, BB, 1024, 256, 0);
    k_sample<<<(BB * CC + 255) / 256, 256, 0, stream>>>(posts + (size_t)t * BB * 1024,
                                                        gum, t, stoch);
    k_write_full<<<(BB * FS + 255) / 256, 256, 0, stream>>>(h, stoch, t, fulls);
  }

  // ---- decoder ----
  gemm(fulls, FS, w_declw, dec_lb, bufC, 1024, NIMG_DEC, 1024, FS, 0);
  convT(bufC, w_dw1, db1, bufA, NIMG_DEC, 1024,  1,  1, 128, 5,  5,  5, 1);
  convT(bufA, w_dw2, db2, bufB, NIMG_DEC,  128,  5,  5,  64, 5, 13, 13, 1);
  convT(bufB, w_dw3, db3, bufA, NIMG_DEC,   64, 13, 13,  32, 6, 30, 30, 1);
  convT(bufA, w_dw4, db4, bufB, NIMG_DEC,   32, 30, 30,   3, 6, 64, 64, 0);

  // ---- reward head ----
  gemm(fulls, FS, w_rw1, rb1, rtmp, 256, NIMG_DEC, 256, FS, 1);
  gemm(rtmp, 256, w_rw2, rb2, rpred, 1, NIMG_DEC, 1, 256, 0);

  // ---- losses ----
  k_mse_recon<<<2048, 256, 0, stream>>>(bufB, obs, accum);
  k_mse_reward<<<(NIMG_DEC + 255) / 256, 256, 0, stream>>>(rpred, rew, accum);
  k_kl<<<128, 256, 0, stream>>>(priors, posts, accum);
  k_final<<<1, 1, 0, stream>>>(accum, out);
}
